// Attention_66932770341587
// MI455X (gfx1250) — matrix operand order repaired and checked by reading, not compiled, not run on hardware
//
#include <hip/hip_runtime.h>
#include <hip/hip_bf16.h>
#include <stdint.h>

#define B_    8
#define L_    2048
#define DIM_  768
#define ZDIM_ 192
#define HDIM_ 768
#define NDIM_ 16
#define MTOT  (B_ * L_)      // 16384
#define NCHUNK 16
#define CLEN   (L_ / NCHUNK) // 128

typedef __attribute__((ext_vector_type(16))) __bf16 v16bf;
typedef __attribute__((ext_vector_type(8)))  __bf16 v8bf;
typedef __attribute__((ext_vector_type(8)))  float  v8f;

static __device__ __forceinline__ float sigm(float x) { return 1.0f / (1.0f + __expf(-x)); }
static __device__ __forceinline__ float silu(float x) { return x * sigm(x); }

static __device__ __forceinline__ unsigned short f2bf(float f) {
  union { float f; uint32_t u; } c; c.f = f;
  uint32_t u = c.u;
  u += 0x7FFFu + ((u >> 16) & 1u);   // round-to-nearest-even
  return (unsigned short)(u >> 16);
}

// ---------------------------------------------------------------- converts
__global__ void f32_to_bf16_kernel(const float* __restrict__ src,
                                   unsigned short* __restrict__ dst, long n) {
  long i = (long)blockIdx.x * blockDim.x + threadIdx.x;
  long stride = (long)gridDim.x * blockDim.x;
  for (; i < n; i += stride) dst[i] = f2bf(src[i]);
}

// ---------------------------------------------------------------- EMA scan
// phase A: per (b,d,chunk) scan from zero state, record 16 final states
__global__ void ema_phaseA(const float* __restrict__ x,
                           const float* __restrict__ delta,
                           const float* __restrict__ alpha,
                           float* __restrict__ sfin) {
  int t = blockIdx.x * blockDim.x + threadIdx.x;   // B*NCHUNK*DIM threads
  int d = t % DIM_;
  int c = (t / DIM_) % NCHUNK;
  int b = t / (DIM_ * NCHUNK);
  float q[NDIM_], s[NDIM_];
#pragma unroll
  for (int n = 0; n < NDIM_; n++) {
    float p = sigm(delta[d * NDIM_ + n]);
    q[n] = 1.0f - p * sigm(alpha[d * NDIM_ + n]);
    s[n] = 0.0f;
  }
  const float* xp = x + ((size_t)b * L_ + (size_t)c * CLEN) * DIM_ + d;
  for (int tt = 0; tt < CLEN; tt++) {
    float xv = xp[(size_t)tt * DIM_];
#pragma unroll
    for (int n = 0; n < NDIM_; n++) s[n] = q[n] * s[n] + xv;
  }
  float* sp = sfin + (((size_t)b * NCHUNK + c) * NDIM_) * DIM_ + d;
#pragma unroll
  for (int n = 0; n < NDIM_; n++) sp[(size_t)n * DIM_] = s[n];
}

// phase B: build carry from previous chunks, rescan, write mx = silu(conv + x*omega) as bf16
__global__ void ema_phaseB(const float* __restrict__ x,
                           const float* __restrict__ delta,
                           const float* __restrict__ alpha,
                           const float* __restrict__ beta,
                           const float* __restrict__ gamma,
                           const float* __restrict__ omega,
                           const float* __restrict__ sfin,
                           unsigned short* __restrict__ mx_bf) {
  int t = blockIdx.x * blockDim.x + threadIdx.x;
  int d = t % DIM_;
  int c = (t / DIM_) % NCHUNK;
  int b = t / (DIM_ * NCHUNK);
  float q[NDIM_], w[NDIM_], qc[NDIM_], s[NDIM_], fac[NDIM_];
  const float scale = 0.25f;   // 1/sqrt(16)
#pragma unroll
  for (int n = 0; n < NDIM_; n++) {
    float p = sigm(delta[d * NDIM_ + n]);
    q[n] = 1.0f - p * sigm(alpha[d * NDIM_ + n]);
    w[n] = p * beta[d * NDIM_ + n] * gamma[d * NDIM_ + n] * scale;
    float qq = q[n];
#pragma unroll
    for (int i = 0; i < 7; i++) qq = qq * qq;     // q^128 = q^CLEN
    qc[n] = qq;
    s[n] = 0.0f;
    fac[n] = 1.0f;
  }
  for (int cp = c - 1; cp >= 0; cp--) {
    const float* sp = sfin + (((size_t)b * NCHUNK + cp) * NDIM_) * DIM_ + d;
#pragma unroll
    for (int n = 0; n < NDIM_; n++) {
      s[n] += sp[(size_t)n * DIM_] * fac[n];
      fac[n] *= qc[n];
    }
  }
  float om = omega[d];
  const size_t base = ((size_t)b * L_ + (size_t)c * CLEN) * DIM_ + d;
  for (int tt = 0; tt < CLEN; tt++) {
    float xv = x[base + (size_t)tt * DIM_];
    float conv = 0.0f;
#pragma unroll
    for (int n = 0; n < NDIM_; n++) {
      s[n] = q[n] * s[n] + xv;
      conv += w[n] * s[n];
    }
    float mv = conv + xv * om;
    mx_bf[base + (size_t)tt * DIM_] = f2bf(silu(mv));
  }
}

// ---------------------------------------------------------------- WMMA GEMM
// out[M,768] = act( A[M,768] @ W[768(N),768(K)]^T + bias )  (bf16 in, f32 acc)
// modes: 0=silu, 1=sigmoid, 2=raw, 3=final (h=silu(val+hx); out=x+u*(h-x))
static __device__ __forceinline__ void store_tile(
    const v8f acc, int r0, int c0, int lane,
    const float* __restrict__ bias, float* __restrict__ out, int mode,
    const float* __restrict__ hx, const float* __restrict__ u,
    const float* __restrict__ xres) {
  const int rr = r0 + 8 * (lane >> 4);
  const int cc = c0 + (lane & 15);
  const float bsv = bias[cc];
#pragma unroll
  for (int e = 0; e < 8; e++) {
    size_t idx = (size_t)(rr + e) * 768 + cc;
    float val = acc[e] + bsv;
    if (mode == 0)      val = silu(val);
    else if (mode == 1) val = sigm(val);
    else if (mode == 3) {
      float hval = silu(val + hx[idx]);
      float xv = xres[idx];
      val = xv + u[idx] * (hval - xv);
    }
    out[idx] = val;
  }
}

// A fragment: per-lane elems 0..7 <- K[koff..koff+7], elems 8..15 <- K[koff+16..koff+23]
static __device__ __forceinline__ v16bf load_fragA(const __bf16* __restrict__ base, int k0) {
  v8bf lo = *(const v8bf*)(base + k0);
  v8bf hi = *(const v8bf*)(base + k0 + 16);
  return __builtin_shufflevector(lo, hi, 0,1,2,3,4,5,6,7,8,9,10,11,12,13,14,15);
}
// B fragment: per-lane elems 0..7 <- K[koff..koff+7], elems 8..15 <- K[koff+16..koff+23] of column n = lane&15
static __device__ __forceinline__ v16bf load_fragB(const __bf16* __restrict__ base, int k0) {
  v8bf lo = *(const v8bf*)(base + k0);
  v8bf hi = *(const v8bf*)(base + k0 + 16);
  return __builtin_shufflevector(lo, hi, 0,1,2,3,4,5,6,7,8,9,10,11,12,13,14,15);
}

__global__ __launch_bounds__(256)
void gemm_bf16_wmma(const __bf16* __restrict__ A,   // M x 768, row-major (K contig)
                    const __bf16* __restrict__ W,   // 768(N) x 768(K), row-major
                    const float*  __restrict__ bias,
                    float*        __restrict__ out,
                    int mode,
                    const float* __restrict__ hx,
                    const float* __restrict__ u,
                    const float* __restrict__ xres) {
  const int K = DIM_;
  const int lane = threadIdx.x & 31;
  const int wave = threadIdx.x >> 5;          // 8 waves
  const int wm = wave & 3;                    // 4 waves along M
  const int wn = wave >> 2;                   // 2 waves along N
  const int row0 = blockIdx.y * 128 + wm * 32;
  const int col0 = blockIdx.x * 128 + wn * 64;

  const int mrow  = lane & 15;
  const int ahalf = (lane >> 4) ? 8  : 0;     // A: K-offset of this lane-half
  const int bhalf = (lane >> 4) ? 8  : 0;     // B: K-offset of this lane-half

  v8f acc[2][4];
#pragma unroll
  for (int mi = 0; mi < 2; mi++)
#pragma unroll
    for (int ni = 0; ni < 4; ni++) acc[mi][ni] = (v8f){};

  const __bf16* Ab[2];
  const __bf16* Wb[4];
#pragma unroll
  for (int mi = 0; mi < 2; mi++)
    Ab[mi] = A + (size_t)(row0 + mi * 16 + mrow) * K + ahalf;
#pragma unroll
  for (int ni = 0; ni < 4; ni++)
    Wb[ni] = W + (size_t)(col0 + ni * 16 + mrow) * K + bhalf;

  for (int k0 = 0; k0 < K; k0 += 32) {
    if (k0 + 32 < K) {
      __builtin_prefetch(Ab[0] + k0 + 32, 0, 3);   // global_prefetch_b8 (near scope)
      __builtin_prefetch(Wb[0] + k0 + 32, 0, 3);
      __builtin_prefetch(Wb[2] + k0 + 32, 0, 3);
    }
    v16bf af[2], bf[4];
#pragma unroll
    for (int mi = 0; mi < 2; mi++) af[mi] = load_fragA(Ab[mi], k0);
#pragma unroll
    for (int ni = 0; ni < 4; ni++) bf[ni] = load_fragB(Wb[ni], k0);

#pragma unroll
    for (int mi = 0; mi < 2; mi++)
#pragma unroll
      for (int ni = 0; ni < 4; ni++)
        acc[mi][ni] = __builtin_amdgcn_wmma_f32_16x16x32_bf16(
            false, af[mi], false, bf[ni], (short)0, acc[mi][ni], false, false);
  }

#pragma unroll
  for (int mi = 0; mi < 2; mi++)
#pragma unroll
    for (int ni = 0; ni < 4; ni++)
      store_tile(acc[mi][ni], row0 + mi * 16, col0 + ni * 16, lane,
                 bias, out, mode, hx, u, xres);
}

// ---------------------------------------------------------------- sort + scale
// one block per (b,h): bitonic sort v[b,:,h] in LDS, apply col_descend flip,
// multiply by r[b,:,h], emit bf16 A-matrix for the final GEMM.
__global__ __launch_bounds__(256)
void sort_scale_kernel(const float* __restrict__ v,
                       const float* __restrict__ r,
                       const unsigned char* __restrict__ col_desc,
                       unsigned short* __restrict__ outA) {
  __shared__ float sv[L_];
  const int b = blockIdx.x / HDIM_;
  const int h = blockIdx.x % HDIM_;
  const int tid = threadIdx.x;
  for (int i = tid; i < L_; i += 256)
    sv[i] = v[((size_t)b * L_ + i) * HDIM_ + h];
  __syncthreads();
  for (int k = 2; k <= L_; k <<= 1) {
    for (int j = k >> 1; j > 0; j >>= 1) {
      for (int i = tid; i < L_; i += 256) {
        int ixj = i ^ j;
        if (ixj > i) {
          bool up = ((i & k) == 0);
          float a = sv[i], c = sv[ixj];
          if ((a > c) == up) { sv[i] = c; sv[ixj] = a; }
        }
      }
      __syncthreads();
    }
  }
  const bool desc = col_desc[h] != 0;
  for (int i = tid; i < L_; i += 256) {
    float val = desc ? sv[L_ - 1 - i] : sv[i];
    size_t idx = ((size_t)b * L_ + i) * HDIM_ + h;
    outA[idx] = f2bf(val * r[idx]);
  }
}

// ---------------------------------------------------------------- launcher
extern "C" void kernel_launch(void* const* d_in, const int* in_sizes, int n_in,
                              void* d_out, int out_size, void* d_ws, size_t ws_size,
                              hipStream_t stream) {
  (void)in_sizes; (void)n_in; (void)out_size; (void)ws_size;
  const float* x        = (const float*)d_in[0];
  const float* vproj_w  = (const float*)d_in[1];
  const float* vproj_b  = (const float*)d_in[2];
  const float* mxproj_w = (const float*)d_in[3];
  const float* mxproj_b = (const float*)d_in[4];
  const float* hproj_w  = (const float*)d_in[5];
  const float* hproj_b  = (const float*)d_in[6];
  const float* delta    = (const float*)d_in[7];
  const float* alpha    = (const float*)d_in[8];
  const float* beta     = (const float*)d_in[9];
  const float* gamma    = (const float*)d_in[10];
  const float* omega    = (const float*)d_in[11];
  const unsigned char* col_desc = (const unsigned char*)d_in[12];
  float* out = (float*)d_out;

  char* ws = (char*)d_ws;
  size_t off = 0;
  auto alloc = [&](size_t bytes) { size_t o = off; off += (bytes + 255) & ~(size_t)255; return (void*)(ws + o); };

  unsigned short* x_bf   = (unsigned short*)alloc((size_t)MTOT * DIM_ * 2);
  unsigned short* mx_bf  = (unsigned short*)alloc((size_t)MTOT * DIM_ * 2);
  unsigned short* srt_bf = (unsigned short*)alloc((size_t)MTOT * HDIM_ * 2);
  unsigned short* vw_bf  = (unsigned short*)alloc((size_t)HDIM_ * DIM_ * 2);
  unsigned short* mxw_bf = (unsigned short*)alloc((size_t)(2 * DIM_ + ZDIM_ + HDIM_) * DIM_ * 2);
  unsigned short* hw_bf  = (unsigned short*)alloc((size_t)DIM_ * HDIM_ * 2);
  float* v_f32  = (float*)alloc((size_t)MTOT * HDIM_ * 4);
  float* u_f32  = (float*)alloc((size_t)MTOT * DIM_ * 4);
  float* r_f32  = (float*)alloc((size_t)MTOT * HDIM_ * 4);
  float* hx_f32 = (float*)alloc((size_t)MTOT * DIM_ * 4);
  float* sfin   = (float*)alloc((size_t)B_ * NCHUNK * NDIM_ * DIM_ * 4);

  // bf16 conversion
  f32_to_bf16_kernel<<<2048, 256, 0, stream>>>(x, x_bf, (long)MTOT * DIM_);
  f32_to_bf16_kernel<<<512,  256, 0, stream>>>(vproj_w, vw_bf, (long)HDIM_ * DIM_);
  f32_to_bf16_kernel<<<1024, 256, 0, stream>>>(mxproj_w, mxw_bf, (long)(2 * DIM_ + ZDIM_ + HDIM_) * DIM_);
  f32_to_bf16_kernel<<<512,  256, 0, stream>>>(hproj_w, hw_bf, (long)DIM_ * HDIM_);

  // EMA chunked scan -> mx (bf16)
  ema_phaseA<<<(B_ * NCHUNK * DIM_) / 256, 256, 0, stream>>>(x, delta, alpha, sfin);
  ema_phaseB<<<(B_ * NCHUNK * DIM_) / 256, 256, 0, stream>>>(x, delta, alpha, beta, gamma, omega, sfin, mx_bf);

  dim3 ggrid(HDIM_ / 128, MTOT / 128);
  // v = silu(x @ vproj_w^T + vb)
  gemm_bf16_wmma<<<ggrid, 256, 0, stream>>>((const __bf16*)x_bf, (const __bf16*)vw_bf,
                                            vproj_b, v_f32, 0, nullptr, nullptr, nullptr);
  // u = sigmoid(base[:, :768]); r = silu(base[:, 960:1728]); hx = base[:, 1728:2496]
  gemm_bf16_wmma<<<ggrid, 256, 0, stream>>>((const __bf16*)mx_bf, (const __bf16*)mxw_bf,
                                            mxproj_b, u_f32, 1, nullptr, nullptr, nullptr);
  gemm_bf16_wmma<<<ggrid, 256, 0, stream>>>((const __bf16*)mx_bf,
                                            (const __bf16*)(mxw_bf + (size_t)(DIM_ + ZDIM_) * DIM_),
                                            mxproj_b + (DIM_ + ZDIM_), r_f32, 0, nullptr, nullptr, nullptr);
  gemm_bf16_wmma<<<ggrid, 256, 0, stream>>>((const __bf16*)mx_bf,
                                            (const __bf16*)(mxw_bf + (size_t)(DIM_ + ZDIM_ + HDIM_) * DIM_),
                                            mxproj_b + (DIM_ + ZDIM_ + HDIM_), hx_f32, 2, nullptr, nullptr, nullptr);

  // sorted(v) flipped per col_descend, times r, as bf16
  sort_scale_kernel<<<B_ * HDIM_, 256, 0, stream>>>(v_f32, r_f32, col_desc, srt_bf);

  // final: h = silu(hx + (sorted*r) @ hproj_w^T + hb); out = x + u*(h - x)
  gemm_bf16_wmma<<<ggrid, 256, 0, stream>>>((const __bf16*)srt_bf, (const __bf16*)hw_bf,
                                            hproj_b, out, 3, hx_f32, u_f32, x);
}
